// RelationalTransformer_24953759990447
// MI455X (gfx1250) — compile-verified, matrix operand order checked by reading, not run on hardware
//
#include <hip/hip_runtime.h>
#include <hip/hip_bf16.h>

#define B_    2
#define S_    1024
#define D_    512
#define H_    8
#define DH_   64
#define DFF_  2048
#define DTXT_ 384
#define M_    (B_ * S_)   /* 2048 token rows */

typedef __attribute__((ext_vector_type(16))) _Float16 v16h;
typedef __attribute__((ext_vector_type(8)))  _Float16 v8h;
typedef __attribute__((ext_vector_type(8)))  float    v8f;
typedef __attribute__((ext_vector_type(4)))  float    f4;

// ---------------------------------------------------------------- WMMA core
__device__ __forceinline__ v8f wmma_f32_16x16x32(v16h a, v16h b, v8f c) {
  // D = A(16x32 f16) * B(32x16 f16) + C(16x16 f32)
  return __builtin_amdgcn_wmma_f32_16x16x32_f16(
      /*neg_a=*/false, a, /*neg_b=*/false, b,
      /*c_mod=*/(short)0, c, /*reuse_a=*/false, /*reuse_b=*/false);
}

// Fragment layout (ISA 16-bit 16x32 A/B):
// lane L: row = L%16 ; hi = L/16 ; element g*8+kk holds K = (hi?8:0)+g*16+kk
struct RawFrag { f4 a, b, c, d; };

__device__ __forceinline__ RawFrag load_raw_f32(const float* rowbase, int hi) {
  const int kb = hi ? 8 : 0;
  RawFrag r;
  r.a = *(const f4*)(rowbase + kb);
  r.b = *(const f4*)(rowbase + kb + 4);
  r.c = *(const f4*)(rowbase + kb + 16);
  r.d = *(const f4*)(rowbase + kb + 20);
  return r;
}
__device__ __forceinline__ v16h cvt_frag(RawFrag r) {
  v16h o;
#pragma unroll
  for (int i = 0; i < 4; ++i) {
    o[i]      = (_Float16)r.a[i];
    o[4 + i]  = (_Float16)r.b[i];
    o[8 + i]  = (_Float16)r.c[i];
    o[12 + i] = (_Float16)r.d[i];
  }
  return o;
}
__device__ __forceinline__ v16h frag_from_h16(const _Float16* rowbase, int hi) {
  const int kb = hi ? 8 : 0;
  v8h lo = *(const v8h*)(rowbase + kb);        // 16B: K kb..kb+7
  v8h up = *(const v8h*)(rowbase + kb + 16);   // 16B: K kb+16..kb+23
  v16h o;
#pragma unroll
  for (int i = 0; i < 8; ++i) { o[i] = lo[i]; o[8 + i] = up[i]; }
  return o;
}

// ------------------------------------------------- CDNA5 async global->LDS
__device__ __forceinline__ unsigned lds_off32(const void* p) {
  return (unsigned)(unsigned long long)p;  // low 32 bits = LDS byte offset
}
__device__ __forceinline__ void async_copy_b128(void* lds_dst, const void* gsrc) {
  unsigned dst = lds_off32(lds_dst);
  unsigned long long src = (unsigned long long)gsrc;
  asm volatile("global_load_async_to_lds_b128 %0, %1, off"
               :: "v"(dst), "v"(src) : "memory");
}
__device__ __forceinline__ void wait_async0() {
  asm volatile("s_wait_asynccnt 0" ::: "memory");
}

// --------------------------------------------------------- lane reductions
__device__ __forceinline__ float hmax16(float x) {
#pragma unroll
  for (int m = 8; m >= 1; m >>= 1) x = fmaxf(x, __shfl_xor(x, m, 16));
  return x;
}
__device__ __forceinline__ float hsum16(float x) {
#pragma unroll
  for (int m = 8; m >= 1; m >>= 1) x += __shfl_xor(x, m, 16);
  return x;
}
__device__ __forceinline__ float wsum32(float x) {
#pragma unroll
  for (int m = 16; m >= 1; m >>= 1) x += __shfl_xor(x, m, 32);
  return x;
}

// ============================================================== GEMM kernel
// C[M,N] = A[M,K] @ W[N,K]^T (+bias) (+= if accum). M%128==0, N%64==0, K%32==0.
// 256 threads = 8 waves; block tile 128x64; wave tile 32x32 (4 WMMA / K-step).
// Double-buffered async global->LDS staging of fp32 tiles.
#define BM 128
#define BN 64
#define KT 32
__global__ __launch_bounds__(256)
void k_gemm(const float* __restrict__ A, const float* __restrict__ W,
            const float* __restrict__ bias, float* __restrict__ C,
            int M, int N, int K, int accum) {
  __shared__ __align__(16) float lA[2][BM * KT];   // 2 x 16 KB
  __shared__ __align__(16) float lW[2][BN * KT];   // 2 x  8 KB
  const int tid  = threadIdx.x;
  const int wave = tid >> 5, lane = tid & 31;
  const int wm = wave >> 1, wn = wave & 1;         // 4 x 2 wave grid
  const int hi = (lane >> 4) & 1;
  const int rowBase = blockIdx.y * BM;
  const int colBase = blockIdx.x * BN;

  v8f acc00 = {}, acc01 = {}, acc10 = {}, acc11 = {};

  // branch-free staging: A tile = 1024 float4 chunks, W tile = 512 chunks
#define PREFETCH(buf, k0)                                                      \
  {                                                                            \
    _Pragma("unroll")                                                          \
    for (int it = 0; it < 4; ++it) {                                           \
      const int i = tid + it * 256;                                            \
      const int r = i >> 3, c4 = (i & 7) * 4;                                  \
      async_copy_b128(&lA[buf][r * KT + c4],                                   \
                      A + (size_t)(rowBase + r) * K + (k0) + c4);              \
    }                                                                          \
    _Pragma("unroll")                                                          \
    for (int it = 0; it < 2; ++it) {                                           \
      const int i = tid + it * 256;                                            \
      const int r = i >> 3, c4 = (i & 7) * 4;                                  \
      async_copy_b128(&lW[buf][r * KT + c4],                                   \
                      W + (size_t)(colBase + r) * K + (k0) + c4);              \
    }                                                                          \
  }

  const int nk = K / KT;
  PREFETCH(0, 0);
  for (int kt = 0; kt < nk; ++kt) {
    wait_async0();       // own async copies for buf kt&1 have landed
    __syncthreads();     // everyone's copies landed; prior reads retired
    if (kt + 1 < nk) PREFETCH((kt + 1) & 1, (kt + 1) * KT);

    const float* bufA = &lA[kt & 1][0];
    const float* bufW = &lW[kt & 1][0];
    // batch all raw LDS b128 loads, then convert, then 4 WMMAs
    RawFrag rA0 = load_raw_f32(bufA + (wm * 32 +      (lane & 15)) * KT, hi);
    RawFrag rA1 = load_raw_f32(bufA + (wm * 32 + 16 + (lane & 15)) * KT, hi);
    RawFrag rB0 = load_raw_f32(bufW + (wn * 32 +      (lane & 15)) * KT, hi);
    RawFrag rB1 = load_raw_f32(bufW + (wn * 32 + 16 + (lane & 15)) * KT, hi);
    v16h a0 = cvt_frag(rA0), a1 = cvt_frag(rA1);
    v16h b0 = cvt_frag(rB0), b1 = cvt_frag(rB1);
    acc00 = wmma_f32_16x16x32(a0, b0, acc00);
    acc01 = wmma_f32_16x16x32(a0, b1, acc01);
    acc10 = wmma_f32_16x16x32(a1, b0, acc10);
    acc11 = wmma_f32_16x16x32(a1, b1, acc11);
  }
#undef PREFETCH

  // C layout: lane L, vgpr j -> row = j + (L<16?0:8), col = L%16
  const int n0  = colBase + wn * 32 + (lane & 15);
  const int mb0 = rowBase + wm * 32 + hi * 8;
  const float biaL = bias ? bias[n0]      : 0.f;
  const float biaH = bias ? bias[n0 + 16] : 0.f;
#pragma unroll
  for (int j = 0; j < 8; ++j) {
    const size_t oT = (size_t)(mb0 + j) * N + n0;       // rows wm*32 + 0..15
    const size_t oB = (size_t)(mb0 + 16 + j) * N + n0;  // rows wm*32 + 16..31
    float v00 = acc00[j] + biaL, v01 = acc01[j] + biaH;
    float v10 = acc10[j] + biaL, v11 = acc11[j] + biaH;
    if (accum) {
      C[oT] += v00; C[oT + 16] += v01;
      C[oB] += v10; C[oB + 16] += v11;
    } else {
      C[oT] = v00; C[oT + 16] = v01;
      C[oB] = v10; C[oB + 16] = v11;
    }
  }
}

// ========================================================= Flash attention
// One wave per (b, head, 16-row Q tile); KV tiles of 32 keys; dh=64.
__global__ __launch_bounds__(256)
void k_attn(const float* __restrict__ Q, const float* __restrict__ Kx,
            const float* __restrict__ V, const unsigned char* __restrict__ msk,
            float* __restrict__ O) {
  __shared__ __align__(16) _Float16 sV[8][DH_ * 32]; // per-wave [d][key]
  __shared__ __align__(16) _Float16 sP[8][16 * 32];  // per-wave [row][key]
  const int wave = threadIdx.x >> 5, lane = threadIdx.x & 31;
  const int hi = (lane >> 4) & 1;
  const int gw = blockIdx.x * 8 + wave;     // 0..1023
  const int qt = gw & 63;
  const int h  = (gw >> 6) & 7;
  const int b  = gw >> 9;
  _Float16* sVw = &sV[wave][0];
  _Float16* sPw = &sP[wave][0];

  const float* qrowp =
      Q + ((size_t)(b * S_ + qt * 16 + (lane & 15))) * D_ + h * DH_;
  RawFrag rq0 = load_raw_f32(qrowp, hi);
  RawFrag rq1 = load_raw_f32(qrowp + 32, hi);
  const v16h qf0 = cvt_frag(rq0), qf1 = cvt_frag(rq1);

  v8f acc[4] = {{}, {}, {}, {}};
  float mrow[8], lrow[8], arow[8];
#pragma unroll
  for (int j = 0; j < 8; ++j) { mrow[j] = -1e30f; lrow[j] = 0.f; arow[j] = 0.f; }

  const float scale = 0.125f;  // 1/sqrt(64)
  const size_t mb0 = (size_t)b * S_ * S_;

  for (int kv0 = 0; kv0 < S_; kv0 += 32) {
    // stage V tile [key 0..31][d 0..63] transposed into LDS as [d][key]
#pragma unroll 4
    for (int i = lane; i < 32 * DH_; i += 32) {
      const int key = i >> 6, d = i & 63;
      sVw[d * 32 + key] =
          (_Float16)V[((size_t)(b * S_ + kv0 + key)) * D_ + h * DH_ + d];
    }
    // scores for two 16-key subtiles (batch raw global loads, then convert)
    const float* kp0 =
        Kx + ((size_t)(b * S_ + kv0 + (lane & 15))) * D_ + h * DH_;
    const float* kp1 = kp0 + (size_t)16 * D_;
    RawFrag rk00 = load_raw_f32(kp0, hi), rk01 = load_raw_f32(kp0 + 32, hi);
    RawFrag rk10 = load_raw_f32(kp1, hi), rk11 = load_raw_f32(kp1 + 32, hi);
    v8f s[2];
    {
      v8f sc = {};
      sc = wmma_f32_16x16x32(qf0, cvt_frag(rk00), sc);
      sc = wmma_f32_16x16x32(qf1, cvt_frag(rk01), sc);
      s[0] = sc;
      v8f sd = {};
      sd = wmma_f32_16x16x32(qf0, cvt_frag(rk10), sd);
      sd = wmma_f32_16x16x32(qf1, cvt_frag(rk11), sd);
      s[1] = sd;
    }
    // masked online softmax; score row = j + hi*8, col = lane%16 (+16 sub 1)
#pragma unroll
    for (int j = 0; j < 8; ++j) {
      const int qrow = qt * 16 + hi * 8 + j;
      const size_t mrb = mb0 + (size_t)qrow * S_ + kv0 + (lane & 15);
      const unsigned char m0 = msk[mrb];
      const unsigned char m1 = msk[mrb + 16];
      const float a0 = m0 ? s[0][j] * scale : -1e30f;
      const float a1 = m1 ? s[1][j] * scale : -1e30f;
      const float anyv = (m0 | m1) ? 1.f : 0.f;
      const float mnew = fmaxf(mrow[j], hmax16(fmaxf(a0, a1)));
      const float p0 = __expf(a0 - mnew);
      const float p1 = __expf(a1 - mnew);
      const float fac = __expf(mrow[j] - mnew);
      lrow[j] = lrow[j] * fac + hsum16(p0 + p1);
      mrow[j] = mnew;
      arow[j] = fmaxf(arow[j], hmax16(anyv));
#pragma unroll
      for (int c = 0; c < 4; ++c) acc[c][j] *= fac;
      sPw[(hi * 8 + j) * 32 + (lane & 15)]      = (_Float16)p0;
      sPw[(hi * 8 + j) * 32 + 16 + (lane & 15)] = (_Float16)p1;
    }
    // P(16x32) @ V(32x64): A-frag from sP, 4 B-frags from sV (b128 LDS reads)
    v16h pf = frag_from_h16(sPw + (lane & 15) * 32, hi);
#pragma unroll
    for (int c = 0; c < 4; ++c) {
      v16h vf = frag_from_h16(sVw + (c * 16 + (lane & 15)) * 32, hi);
      acc[c] = wmma_f32_16x16x32(pf, vf, acc[c]);
    }
  }

  float inv[8];
#pragma unroll
  for (int j = 0; j < 8; ++j)
    inv[j] = (arow[j] > 0.f && lrow[j] > 0.f) ? 1.f / lrow[j] : 0.f;
#pragma unroll
  for (int c = 0; c < 4; ++c)
#pragma unroll
    for (int j = 0; j < 8; ++j) {
      const int qrow = qt * 16 + hi * 8 + j;
      const int d = c * 16 + (lane & 15);
      O[((size_t)(b * S_ + qrow)) * D_ + h * DH_ + d] = acc[c][j] * inv[j];
    }
}

// ============================================================ row kernels
__global__ __launch_bounds__(256)
void k_rmsnorm(const float* __restrict__ x, const float* __restrict__ g,
               float* __restrict__ o) {
  const int wave = threadIdx.x >> 5, lane = threadIdx.x & 31;
  const int row = blockIdx.x * 8 + wave;
  const float* xr = x + (size_t)row * D_;
  float v[16], ss = 0.f;
#pragma unroll
  for (int i = 0; i < 16; ++i) { float a = xr[lane + 32 * i]; v[i] = a; ss += a * a; }
  ss = wsum32(ss);
  const float r = rsqrtf(ss * (1.f / D_) + 1e-6f);
  float* orow = o + (size_t)row * D_;
#pragma unroll
  for (int i = 0; i < 16; ++i) { int d = lane + 32 * i; orow[d] = v[i] * r * g[d]; }
}

__device__ __forceinline__ float rms_of(float ss) {
  return rsqrtf(ss * (1.f / D_) + 1e-6f);
}

__global__ __launch_bounds__(256)
void k_embed(const float* __restrict__ numv, const float* __restrict__ dtv,
             const float* __restrict__ bov, const float* __restrict__ tproj,
             const float* __restrict__ cnproj, const int* __restrict__ st,
             const unsigned char* __restrict__ msk,
             const float* nw, const float* nb, const float* num_g,
             const float* dw, const float* db, const float* dt_g,
             const float* bw, const float* bb, const float* bool_g,
             const float* text_g, const float* cn_g,
             const float* mask_emb, float* __restrict__ h) {
  const int wave = threadIdx.x >> 5, lane = threadIdx.x & 31;
  const int row = blockIdx.x * 8 + wave;
  const float xn = numv[row], xd = dtv[row], xb = bov[row];
  float vn[16], vd[16], vb_[16], vt[16], vc[16];
  float sn = 0, sd = 0, sb = 0, stx = 0, sc = 0;
#pragma unroll
  for (int i = 0; i < 16; ++i) {
    const int d = lane + 32 * i;
    float a;
    a = xn * nw[d] + nb[d]; vn[i] = a; sn += a * a;
    a = xd * dw[d] + db[d]; vd[i] = a; sd += a * a;
    a = xb * bw[d] + bb[d]; vb_[i] = a; sb += a * a;
    a = tproj[(size_t)row * D_ + d];  vt[i] = a; stx += a * a;
    a = cnproj[(size_t)row * D_ + d]; vc[i] = a; sc += a * a;
  }
  const float rn = rms_of(wsum32(sn)), rd = rms_of(wsum32(sd));
  const float rb = rms_of(wsum32(sb)), rt = rms_of(wsum32(stx));
  const float rc = rms_of(wsum32(sc));
  const int t = st[row];
  const int mk = msk[row];
#pragma unroll
  for (int i = 0; i < 16; ++i) {
    const int d = lane + 32 * i;
    const float ne = vn[i] * rn * num_g[d];
    const float te = vt[i] * rt * text_g[d];
    const float de = vd[i] * rd * dt_g[d];
    const float be = vb_[i] * rb * bool_g[d];
    float val = (t == 0) ? ne : (t == 1) ? te : (t == 2) ? de : be;
    if (mk) val = mask_emb[t * D_ + d];
    h[(size_t)row * D_ + d] = vc[i] * rc * cn_g[d] + val;
  }
}

__global__ __launch_bounds__(256)
void k_silu_mul(float* __restrict__ g, const float* __restrict__ u, int n4) {
  for (int i = blockIdx.x * 256 + threadIdx.x; i < n4; i += gridDim.x * 256) {
    f4 x = ((const f4*)g)[i];
    f4 y = ((const f4*)u)[i];
    f4 r;
#pragma unroll
    for (int c = 0; c < 4; ++c) r[c] = (x[c] / (1.f + __expf(-x[c]))) * y[c];
    ((f4*)g)[i] = r;
  }
}

__global__ __launch_bounds__(256)
void k_decode(const float* __restrict__ xn, const float* __restrict__ ttxt,
              const float* nw, const float* nb, const float* dw, const float* db,
              const float* bw, const float* bb, float* __restrict__ out) {
  const int wave = threadIdx.x >> 5, lane = threadIdx.x & 31;
  const int row = blockIdx.x * 8 + wave;
  float pn = 0, pd = 0, pb = 0;
#pragma unroll
  for (int i = 0; i < 16; ++i) {
    const int d = lane + 32 * i;
    const float x = xn[(size_t)row * D_ + d];
    pn += x * nw[d]; pd += x * dw[d]; pb += x * bw[d];
  }
  pn = wsum32(pn); pd = wsum32(pd); pb = wsum32(pb);
  float* orow = out + (size_t)row * (DTXT_ + 3);
  if (lane == 0) { orow[0] = pn + nb[0]; orow[1] = pd + db[0]; orow[2] = pb + bb[0]; }
  for (int j = lane; j < DTXT_; j += 32)
    orow[3 + j] = ttxt[(size_t)row * DTXT_ + j];
}

// ================================================================= host
// Flattened input layout (jax tree-flatten, sorted dict keys):
enum {
  IN_NUM = 0, IN_DT, IN_BOOL, IN_TEXT, IN_CN, IN_ST, IN_MASKS, IN_PAD,
  IN_M_COL, IN_M_FEAT, IN_M_NB, IN_M_FULL,
  P_BB = 12, P_BLOCKS = 13,  // 4 blocks x 24 leaves
  P_BOOL_G = 109, P_BW, P_CN_G, P_CNB, P_CNW, P_DB,
  P_DEC_BB, P_DEC_BW, P_DEC_DB, P_DEC_DW, P_DEC_NB, P_DEC_NW,
  P_DEC_TB, P_DEC_TW, P_DT_G, P_DW, P_MASK_EMB, P_NB, P_NUM_G, P_NW,
  P_OUT_G, P_TB, P_TEXT_G, P_TW
};
// per-block leaf offsets (sorted): attn_col{wk,wo,wq,wv}=0..3, attn_feat=4..7,
// attn_full=8..11, attn_nb=12..15, ffn{down,gate,up}=16..18,
// norm_col=19, norm_feat=20, norm_ffn=21, norm_full=22, norm_nb=23

extern "C" void kernel_launch(void* const* d_in, const int* in_sizes, int n_in,
                              void* d_out, int out_size, void* d_ws, size_t ws_size,
                              hipStream_t stream) {
  (void)in_sizes; (void)n_in; (void)out_size; (void)ws_size;
  auto F = [&](int i) -> const float* { return (const float*)d_in[i]; };
  auto U8 = [&](int i) -> const unsigned char* { return (const unsigned char*)d_in[i]; };

  float* ws = (float*)d_ws;
  float* h  = ws;
  float* xn = h  + (size_t)M_ * D_;
  float* qb = xn + (size_t)M_ * D_;
  float* kb = qb + (size_t)M_ * D_;
  float* vb = kb + (size_t)M_ * D_;
  float* ao = vb + (size_t)M_ * D_;
  float* gb = ao + (size_t)M_ * D_;
  float* ub = gb + (size_t)M_ * DFF_;

  auto gemm = [&](const float* A, const float* W, const float* bias, float* C,
                  int N, int K, int accum) {
    dim3 grid(N / 64, M_ / 128);
    k_gemm<<<grid, 256, 0, stream>>>(A, W, bias, C, M_, N, K, accum);
  };
  auto rms = [&](const float* x, const float* g, float* o) {
    k_rmsnorm<<<M_ / 8, 256, 0, stream>>>(x, g, o);
  };

  // ---- embeddings: 384->512 projections, then fused embed
  gemm(F(IN_CN),   F(P_CNW), F(P_CNB), ao, D_, DTXT_, 0);  // column-name proj
  gemm(F(IN_TEXT), F(P_TW),  F(P_TB),  gb, D_, DTXT_, 0);  // text proj
  k_embed<<<M_ / 8, 256, 0, stream>>>(
      F(IN_NUM), F(IN_DT), F(IN_BOOL), gb, ao,
      (const int*)d_in[IN_ST], U8(IN_MASKS),
      F(P_NW), F(P_NB), F(P_NUM_G),
      F(P_DW), F(P_DB), F(P_DT_G),
      F(P_BW), F(P_BB), F(P_BOOL_G),
      F(P_TEXT_G), F(P_CN_G), F(P_MASK_EMB), h);

  // ---- transformer blocks
  const int wOff[4] = {0, 4, 12, 8};              // col, feat, nb, full
  const int nOff[4] = {19, 20, 23, 22};
  const int mIdx[4] = {IN_M_COL, IN_M_FEAT, IN_M_NB, IN_M_FULL};
  for (int blk = 0; blk < 4; ++blk) {
    const int base = P_BLOCKS + 24 * blk;
    for (int a = 0; a < 4; ++a) {
      const float* wk = F(base + wOff[a] + 0);
      const float* wo = F(base + wOff[a] + 1);
      const float* wq = F(base + wOff[a] + 2);
      const float* wv = F(base + wOff[a] + 3);
      rms(h, F(base + nOff[a]), xn);
      gemm(xn, wq, nullptr, qb, D_, D_, 0);
      gemm(xn, wk, nullptr, kb, D_, D_, 0);
      gemm(xn, wv, nullptr, vb, D_, D_, 0);
      k_attn<<<(B_ * H_ * (S_ / 16)) / 8, 256, 0, stream>>>(
          qb, kb, vb, U8(mIdx[a]), ao);
      gemm(ao, wo, nullptr, h, D_, D_, 1);        // h += attn @ wo^T
    }
    // FFN
    rms(h, F(base + 21), xn);
    gemm(xn, F(base + 17), nullptr, gb, DFF_, D_, 0);   // gate
    gemm(xn, F(base + 18), nullptr, ub, DFF_, D_, 0);   // up
    k_silu_mul<<<1024, 256, 0, stream>>>(gb, ub, (M_ * DFF_) / 4);
    gemm(gb, F(base + 16), nullptr, h, D_, DFF_, 1);    // h += . @ down^T
  }

  // ---- decode
  rms(h, F(P_OUT_G), xn);
  gemm(xn, F(P_DEC_TW), F(P_DEC_TB), qb, DTXT_, D_, 0); // text head -> qb
  k_decode<<<M_ / 8, 256, 0, stream>>>(
      xn, qb, F(P_DEC_NW), F(P_DEC_NB), F(P_DEC_DW), F(P_DEC_DB),
      F(P_DEC_BW), F(P_DEC_BB), (float*)d_out);
}
